// BahdanauAttention_19662360281167
// MI455X (gfx1250) — matrix operand order repaired and checked by reading, not compiled, not run on hardware
//
#include <hip/hip_runtime.h>
#include <hip/hip_bf16.h>
#include <math.h>

// ---------------------------------------------------------------------------
// Bahdanau attention for MI455X (gfx1250, wave32, WMMA).
//
// Dominant cost: pk = key[8192,2048] @ Wk[2048,2048]^T  (68.7 GFLOP).
// Compute-bound (≈860 flop/byte) -> v_wmma_f32_16x16x32_bf16 (f32 inputs
// converted to bf16 in LDS staging, f32 accumulation), 128x128 output tile
// per WG. K streamed in 64-wide chunks with:
//   - one-deep global->register pipeline (HBM/L2 latency behind WMMA loop)
//   - one-deep LDS->register A-fragment rotation (DS latency behind WMMA)
// ---------------------------------------------------------------------------

typedef __attribute__((ext_vector_type(16))) __bf16 v16bf;
typedef __attribute__((ext_vector_type(8)))  __bf16 v8bf;
typedef __attribute__((ext_vector_type(8)))  float  v8f;

#define HDIM 2048
#define KDIM 2048
#define SDIM 8192
#define VDIM 2048

#define TILE_S 128
#define TILE_H 128
#define KCHUNK 64
#define LPITCH 72   // bf16 elems per LDS row: 64 + 8 pad (144B rows, 16B aligned)

// workspace layout (float offsets)
#define WS_PQ 0                       // pq[2048]
#define WS_SP 2048                    // scorePartial[S][16]
#define WS_CP (2048 + SDIM * 16)      // ctxPartial[32][V]
// total = 2048 + 131072 + 65536 floats = 794,624 B

// ---------------------------------------------------------------------------
// Kernel 1: pq[h] = sum_q query[q] * Wq[h,q]   (16 MB stream, ~0.7us)
// ---------------------------------------------------------------------------
__global__ __launch_bounds__(256) void pq_kernel(
    const float* __restrict__ query, const float* __restrict__ Wq,
    float* __restrict__ pq)
{
    const int h = blockIdx.x * 256 + threadIdx.x;
    const float4* q4 = (const float4*)query;
    const float4* w4 = (const float4*)(Wq + (size_t)h * KDIM);
    float acc = 0.f;
    for (int i = 0; i < KDIM / 4; ++i) {
        float4 a = q4[i], b = w4[i];
        acc += a.x * b.x + a.y * b.y + a.z * b.z + a.w * b.w;
    }
    pq[h] = acc;
}

// ---------------------------------------------------------------------------
// Kernel 2: fused pk GEMM (bf16 WMMA) + tanh(pq+pk)·Ws partial-score reduce.
// Grid: (H/TILE_H = 16, S/TILE_S = 64), 256 threads = 8 waves.
// Wave w owns h-subtile w (16 cols) and all 8 s-subtiles (8 C tiles / wave).
// ---------------------------------------------------------------------------
__global__ __launch_bounds__(256) void pk_scores_kernel(
    const float* __restrict__ key, const float* __restrict__ Wk,
    const float* __restrict__ Ws, const float* __restrict__ pq,
    float* __restrict__ scorePartial)
{
    __shared__ __align__(16) __bf16 As[TILE_S * LPITCH];  // key tile (bf16)
    __shared__ __align__(16) __bf16 Bs[TILE_H * LPITCH];  // Wk  tile (bf16)
    __shared__ float waveSum[8 * TILE_S];

    const int tid  = threadIdx.x;
    const int wave = tid >> 5;
    const int lane = tid & 31;
    const int n    = lane & 15;
    const bool hi  = lane >= 16;
    const size_t sBase = (size_t)blockIdx.y * TILE_S;
    const size_t hBase = (size_t)blockIdx.x * TILE_H;

    // staging geometry: 2048 float4 slots / 256 threads = 8 each
    const int srow = tid >> 4;          // rows advance by 16 per i-step
    const int scol = (tid & 15) * 4;

    v8f acc[8];
#pragma unroll
    for (int i = 0; i < 8; ++i)
#pragma unroll
        for (int r = 0; r < 8; ++r) acc[i][r] = 0.f;

    // in-flight global data for the one-deep pipeline
    float4 ra[8], rb[8];
#pragma unroll
    for (int i = 0; i < 8; ++i) {
        int row = srow + i * 16;
        ra[i] = *(const float4*)(key + (sBase + row) * KDIM + scol);
        rb[i] = *(const float4*)(Wk  + (hBase + row) * KDIM + scol);
    }

    for (int k0 = 0; k0 < KDIM; k0 += KCHUNK) {
        // ---- convert current chunk's registers -> bf16 LDS ----
#pragma unroll
        for (int i = 0; i < 8; ++i) {
            int row = srow + i * 16;
            __bf16* da = &As[row * LPITCH + scol];
            da[0] = (__bf16)ra[i].x; da[1] = (__bf16)ra[i].y;
            da[2] = (__bf16)ra[i].z; da[3] = (__bf16)ra[i].w;
            __bf16* db = &Bs[row * LPITCH + scol];
            db[0] = (__bf16)rb[i].x; db[1] = (__bf16)rb[i].y;
            db[2] = (__bf16)rb[i].z; db[3] = (__bf16)rb[i].w;
        }
        __syncthreads();

        // ---- issue next chunk's global loads; wait lands after the WMMAs ----
        if (k0 + KCHUNK < KDIM) {
#pragma unroll
            for (int i = 0; i < 8; ++i) {
                int row = srow + i * 16;
                ra[i] = *(const float4*)(key + (sBase + row) * KDIM + k0 + KCHUNK + scol);
                rb[i] = *(const float4*)(Wk  + (hBase + row) * KDIM + k0 + KCHUNK + scol);
            }
        }

        // ---- two K=32 WMMA steps per chunk, A-fragment rotated by one ----
#pragma unroll
        for (int ks = 0; ks < KCHUNK; ks += 32) {
            // B fragment (32x16): two 8-elem runs at k+{0|8} and k+16+{0|8}
            const int bOff = ks + (hi ? 8 : 0);
            const int brow = wave * 16 + n;
            v8bf b0 = *(const v8bf*)(&Bs[brow * LPITCH + bOff]);
            v8bf b1 = *(const v8bf*)(&Bs[brow * LPITCH + bOff + 16]);
            v16bf bf = __builtin_shufflevector(b0, b1,
                0, 1, 2, 3, 4, 5, 6, 7, 8, 9, 10, 11, 12, 13, 14, 15);

            // A fragment (16x32): two 8-elem runs at k+{0|8} and k+16+{0|8}
            const int aOff = ks + (hi ? 8 : 0);
            v8bf a0 = *(const v8bf*)(&As[n * LPITCH + aOff]);
            v8bf a1 = *(const v8bf*)(&As[n * LPITCH + aOff + 16]);
            v16bf afNext = __builtin_shufflevector(a0, a1,
                0, 1, 2, 3, 4, 5, 6, 7, 8, 9, 10, 11, 12, 13, 14, 15);

#pragma unroll
            for (int ms = 0; ms < 8; ++ms) {
                v16bf af = afNext;
                if (ms < 7) {
                    // prefetch next A fragment before issuing this WMMA so the
                    // wait before WMMA(ms+1) tolerates the in-flight DS loads
                    const int arow = (ms + 1) * 16 + n;
                    v8bf p0 = *(const v8bf*)(&As[arow * LPITCH + aOff]);
                    v8bf p1 = *(const v8bf*)(&As[arow * LPITCH + aOff + 16]);
                    afNext = __builtin_shufflevector(p0, p1,
                        0, 1, 2, 3, 4, 5, 6, 7, 8, 9, 10, 11, 12, 13, 14, 15);
                }
                acc[ms] = __builtin_amdgcn_wmma_f32_16x16x32_bf16(
                    false, af, false, bf, (short)0, acc[ms], false, false);
            }
        }
        __syncthreads();
    }

    // ---- epilogue: score_partial[s] += sum_n Ws[h]*tanh(pq[h]+pk[s,h]) ----
    const size_t h = hBase + wave * 16 + n;
    const float pqh = pq[h];
    const float wsh = Ws[h];
#pragma unroll
    for (int ms = 0; ms < 8; ++ms) {
#pragma unroll
        for (int r = 0; r < 8; ++r) {
            float v = tanhf(pqh + acc[ms][r]) * wsh;
            // deterministic reduction over the 16 lanes of each half
            v += __shfl_xor(v, 1, 32);
            v += __shfl_xor(v, 2, 32);
            v += __shfl_xor(v, 4, 32);
            v += __shfl_xor(v, 8, 32);
            if (n == 0)
                waveSum[wave * TILE_S + ms * 16 + r + (hi ? 8 : 0)] = v;
        }
    }
    __syncthreads();
    if (tid < TILE_S) {
        float s = 0.f;
#pragma unroll
        for (int w = 0; w < 8; ++w) s += waveSum[w * TILE_S + tid];
        scorePartial[(sBase + tid) * 16 + blockIdx.x] = s;
    }
}

// ---------------------------------------------------------------------------
// Kernel 3: reduce 16 h-block partials, softmax over S=8192 (single block).
// ---------------------------------------------------------------------------
__global__ __launch_bounds__(256) void softmax_kernel(
    const float* __restrict__ sp, float* __restrict__ attn)
{
    __shared__ float red[256];
    const int tid = threadIdx.x;
    float sc[SDIM / 256];
    float lmax = -3.4e38f;
#pragma unroll
    for (int i = 0; i < SDIM / 256; ++i) {
        int s = i * 256 + tid;
        float v = 0.f;
#pragma unroll
        for (int hb = 0; hb < 16; ++hb) v += sp[s * 16 + hb];
        sc[i] = v;
        lmax = fmaxf(lmax, v);
    }
    red[tid] = lmax; __syncthreads();
    for (int o = 128; o > 0; o >>= 1) {
        if (tid < o) red[tid] = fmaxf(red[tid], red[tid + o]);
        __syncthreads();
    }
    const float gmax = red[0]; __syncthreads();

    float lsum = 0.f;
#pragma unroll
    for (int i = 0; i < SDIM / 256; ++i) { sc[i] = __expf(sc[i] - gmax); lsum += sc[i]; }
    red[tid] = lsum; __syncthreads();
    for (int o = 128; o > 0; o >>= 1) {
        if (tid < o) red[tid] += red[tid + o];
        __syncthreads();
    }
    const float inv = 1.f / red[0];
#pragma unroll
    for (int i = 0; i < SDIM / 256; ++i) attn[i * 256 + tid] = sc[i] * inv;
}

// ---------------------------------------------------------------------------
// Kernel 4/5: context = attn @ values (64 MB stream), split over 32 S-chunks
// for occupancy, then deterministic tree reduce (no float atomics).
// ---------------------------------------------------------------------------
__global__ __launch_bounds__(256) void ctx_partial_kernel(
    const float* __restrict__ attn, const float* __restrict__ values,
    float* __restrict__ cp)
{
    const int v = blockIdx.x * 256 + threadIdx.x;
    const int c = blockIdx.y;                   // 0..31 -> 256 s rows each
    float acc = 0.f;
    for (int i = 0; i < SDIM / 32; ++i) {
        int s = c * (SDIM / 32) + i;
        acc += attn[s] * values[(size_t)s * VDIM + v];
    }
    cp[c * VDIM + v] = acc;
}

__global__ __launch_bounds__(256) void ctx_reduce_kernel(
    const float* __restrict__ cp, float* __restrict__ ctx)
{
    const int v = blockIdx.x * 256 + threadIdx.x;
    float a = 0.f;
#pragma unroll
    for (int c = 0; c < 32; ++c) a += cp[c * VDIM + v];
    ctx[v] = a;
}

// ---------------------------------------------------------------------------
extern "C" void kernel_launch(void* const* d_in, const int* in_sizes, int n_in,
                              void* d_out, int out_size, void* d_ws, size_t ws_size,
                              hipStream_t stream)
{
    const float* query  = (const float*)d_in[0];   // [1,1,2048]
    const float* key    = (const float*)d_in[1];   // [1,8192,2048]
    const float* values = (const float*)d_in[2];   // [8192,2048]
    const float* Wq     = (const float*)d_in[3];   // [2048,2048]
    const float* Wk     = (const float*)d_in[4];   // [2048,2048]
    const float* Ws     = (const float*)d_in[5];   // [1,2048]

    float* ws   = (float*)d_ws;
    float* pq   = ws + WS_PQ;
    float* sp   = ws + WS_SP;
    float* cp   = ws + WS_CP;
    float* out  = (float*)d_out;
    float* attn = out;          // [1,1,8192] first tuple element
    float* ctx  = out + SDIM;   // [1,1,2048] second tuple element

    pq_kernel<<<dim3(HDIM / 256), 256, 0, stream>>>(query, Wq, pq);
    pk_scores_kernel<<<dim3(HDIM / TILE_H, SDIM / TILE_S), 256, 0, stream>>>(
        key, Wk, Ws, pq, sp);
    softmax_kernel<<<1, 256, 0, stream>>>(sp, attn);
    ctx_partial_kernel<<<dim3(VDIM / 256, 32), 256, 0, stream>>>(attn, values, cp);
    ctx_reduce_kernel<<<dim3(VDIM / 256), 256, 0, stream>>>(cp, ctx);
}
